// WanS2VTransformerBlock_30605936951396
// MI455X (gfx1250) — compile-verified, matrix operand order checked by reading, not run on hardware
//
#include <hip/hip_runtime.h>
#include <hip/hip_bf16.h>
#include <math.h>

// ---------------------------------------------------------------------------
// Problem constants (from reference)
// ---------------------------------------------------------------------------
#define DMODEL 1536
#define NHEAD  12
#define DHEAD  128
#define FFNDIM 8960
#define BATCH  2
#define SEQ    2048
#define LENC   512

typedef __attribute__((ext_vector_type(16))) __bf16 v16bf;
typedef __attribute__((ext_vector_type(8)))  float  v8f;

union Frag { v16bf v; unsigned u[8]; };

__device__ __forceinline__ unsigned short f32_to_bf16(float f) {
    union { float f; unsigned u; } v; v.f = f;
    unsigned r = v.u + 0x7FFFu + ((v.u >> 16) & 1u);   // round-to-nearest-even
    return (unsigned short)(r >> 16);
}

// K-pair index pattern for 16-bit WMMA A/B fragments (ISA 7.12.2):
// lanes 0-15 hold K pairs {0..3, 8..11}, lanes 16-31 hold {4..7, 12..15}.
__device__ __forceinline__ int kpair(int v, int half) {
    return (v < 4 ? v : v + 4) + 4 * half;
}

// 16-lane xor-butterfly step. v_permlane16_b32 (pure VALU) when available;
// nibble tables encode sel[i] = i ^ mask within each 16-lane row.
__device__ __forceinline__ float xor16(float x, int mask, unsigned s0, unsigned s1) {
#if __has_builtin(__builtin_amdgcn_permlane16)
    union { float f; unsigned u; } a; a.f = x;
    a.u = __builtin_amdgcn_permlane16(a.u, a.u, s0, s1, false, false);
    return a.f;
#else
    return __shfl_xor(x, mask, 32);
#endif
}
__device__ __forceinline__ float redmax16(float v) {
    v = fmaxf(v, xor16(v, 1, 0x67452301u, 0xEFCDAB89u));
    v = fmaxf(v, xor16(v, 2, 0x54761032u, 0xDCFE98BAu));
    v = fmaxf(v, xor16(v, 4, 0x32107654u, 0xBA98FEDCu));
    v = fmaxf(v, xor16(v, 8, 0xFEDCBA98u, 0x76543210u));
    return v;
}
__device__ __forceinline__ float redsum16(float v) {
    v += xor16(v, 1, 0x67452301u, 0xEFCDAB89u);
    v += xor16(v, 2, 0x54761032u, 0xDCFE98BAu);
    v += xor16(v, 4, 0x32107654u, 0xBA98FEDCu);
    v += xor16(v, 8, 0xFEDCBA98u, 0x76543210u);
    return v;
}

__device__ __forceinline__ float gelu_tanh(float x) {
    float x3 = x * x * x;
    return 0.5f * x * (1.0f + tanhf(0.7978845608028654f * (x + 0.044715f * x3)));
}

// ---------------------------------------------------------------------------
// fp32 -> bf16 conversion (plain, for GEMM A operands)
// ---------------------------------------------------------------------------
__global__ void cvt_f32_bf16(const float* __restrict__ in,
                             unsigned short* __restrict__ out, size_t n) {
    for (size_t i = (size_t)blockIdx.x * blockDim.x + threadIdx.x; i < n;
         i += (size_t)gridDim.x * blockDim.x)
        out[i] = f32_to_bf16(in[i]);
}

// ---------------------------------------------------------------------------
// fp32 (R x C) -> bf16 transposed (C x R).  Tiled 32x32 via LDS, coalesced
// on both sides.  Weights become W^T so GEMM B-fragments are contiguous.
// ---------------------------------------------------------------------------
__global__ void __launch_bounds__(256)
transpose_cvt(const float* __restrict__ in, unsigned short* __restrict__ out,
              int R, int C) {
    __shared__ unsigned short tile[32][33];
    const int tx = threadIdx.x & 31, ty = threadIdx.x >> 5;   // 32 x 8
    const int c0 = blockIdx.x * 32, r0 = blockIdx.y * 32;
#pragma unroll
    for (int j = 0; j < 32; j += 8)
        tile[ty + j][tx] = f32_to_bf16(in[(size_t)(r0 + ty + j) * C + c0 + tx]);
    __syncthreads();
#pragma unroll
    for (int j = 0; j < 32; j += 8)
        out[(size_t)(c0 + ty + j) * R + r0 + tx] = tile[tx][ty + j];
}

// ---------------------------------------------------------------------------
// LayerNorm + AdaLN modulation:  out = LN(x) * (1 + scale_tok) + shift_tok
// ---------------------------------------------------------------------------
__global__ void ln_mod_kernel(const float* __restrict__ x,
                              const float* __restrict__ temb,
                              const float* __restrict__ sst,
                              const int* __restrict__ segp,
                              int shiftIdx, int scaleIdx,
                              unsigned short* __restrict__ out) {
    __shared__ float s1[256], s2[256];
    const int row = blockIdx.x;            // b*SEQ + l
    const int b = row / SEQ, l = row % SEQ;
    const int tid = threadIdx.x;
    float vals[DMODEL / 256];
    float sum = 0.f, sq = 0.f;
#pragma unroll
    for (int j = 0; j < DMODEL / 256; ++j) {
        float v = x[(size_t)row * DMODEL + tid + j * 256];
        vals[j] = v; sum += v; sq += v * v;
    }
    s1[tid] = sum; s2[tid] = sq;
    __syncthreads();
    for (int st = 128; st > 0; st >>= 1) {
        if (tid < st) { s1[tid] += s1[tid + st]; s2[tid] += s2[tid + st]; }
        __syncthreads();
    }
    const float mean = s1[0] * (1.0f / DMODEL);
    const float var  = s2[0] * (1.0f / DMODEL) - mean * mean;
    const float inv  = rsqrtf(var + 1e-6f);
    int seg_idx = segp[0];
    seg_idx = seg_idx < 0 ? 0 : (seg_idx > SEQ ? SEQ : seg_idx);
    const int seg = (l >= seg_idx) ? 1 : 0;
#pragma unroll
    for (int j = 0; j < DMODEL / 256; ++j) {
        int c = tid + j * 256;
        float sc = sst[scaleIdx * DMODEL + c] +
                   temb[(((size_t)b * 6 + scaleIdx) * 2 + seg) * DMODEL + c];
        float sh = sst[shiftIdx * DMODEL + c] +
                   temb[(((size_t)b * 6 + shiftIdx) * 2 + seg) * DMODEL + c];
        float y = (vals[j] - mean) * inv * (1.0f + sc) + sh;
        out[(size_t)row * DMODEL + c] = f32_to_bf16(y);
    }
}

// ---------------------------------------------------------------------------
// Gated residual: out = base + o * gate_tok   (gateIdx < 0 -> plain add)
// ---------------------------------------------------------------------------
__global__ void gate_residual(const float* __restrict__ base,
                              const float* __restrict__ o,
                              const float* __restrict__ temb,
                              const float* __restrict__ sst,
                              const int* __restrict__ segp, int gateIdx,
                              float* __restrict__ out, size_t n) {
    int seg_idx = segp[0];
    seg_idx = seg_idx < 0 ? 0 : (seg_idx > SEQ ? SEQ : seg_idx);
    for (size_t i = (size_t)blockIdx.x * blockDim.x + threadIdx.x; i < n;
         i += (size_t)gridDim.x * blockDim.x) {
        float g = 1.0f;
        if (gateIdx >= 0) {
            int c = (int)(i % DMODEL);
            size_t row = i / DMODEL;
            int l = (int)(row % SEQ), b = (int)(row / SEQ);
            int seg = (l >= seg_idx) ? 1 : 0;
            g = sst[gateIdx * DMODEL + c] +
                temb[(((size_t)b * 6 + gateIdx) * 2 + seg) * DMODEL + c];
        }
        out[i] = base[i] + o[i] * g;
    }
}

// ---------------------------------------------------------------------------
// RoPE on (B,S,D) laid-out heads; f32 in, bf16 out.
// ---------------------------------------------------------------------------
__global__ void rope_bf16(const float* __restrict__ qin,
                          const float* __restrict__ cb,
                          const float* __restrict__ sb,
                          unsigned short* __restrict__ qout, size_t npairs) {
    for (size_t p = (size_t)blockIdx.x * blockDim.x + threadIdx.x; p < npairs;
         p += (size_t)gridDim.x * blockDim.x) {
        size_t row = p / (DMODEL / 2);
        int cp = (int)(p % (DMODEL / 2));
        int i = cp & (DHEAD / 2 - 1);       // (c % 128) / 2
        float cs = cb[row * (DHEAD / 2) + i];
        float sn = sb[row * (DHEAD / 2) + i];
        float xr = qin[row * DMODEL + 2 * cp];
        float xi = qin[row * DMODEL + 2 * cp + 1];
        qout[row * DMODEL + 2 * cp]     = f32_to_bf16(xr * cs - xi * sn);
        qout[row * DMODEL + 2 * cp + 1] = f32_to_bf16(xr * sn + xi * cs);
    }
}

// ---------------------------------------------------------------------------
// bf16 WMMA GEMM with transposed B:  C[M,N] = A[M,K] @ Bt[N,K]^T
// EPI: 0 f32 | 1 bf16 | 2 f32+bias | 3 bf16+bias+gelu
// Block 128x256, BK=64, 8 waves (2 in M x 4 in N), wave tile 64x64 (4x4 WMMA).
// Both A and B fragments load as contiguous ds_load_b128 (same K-pair layout).
// ---------------------------------------------------------------------------
#define BM 128
#define BN 256
#define BK 64

template <int EPI>
__global__ void __launch_bounds__(256)
gemm_nt(const unsigned short* __restrict__ A,
        const unsigned short* __restrict__ Bt,
        const float* __restrict__ bias, void* __restrict__ Cout,
        int M, int N, int K) {
    __shared__ __align__(16) unsigned sA[BM * BK / 2];   // [128][32] dwords
    __shared__ __align__(16) unsigned sB[BN * BK / 2];   // [256][32] dwords
    const int tid  = threadIdx.x;
    const int lane = tid & 31;
    const int wave = tid >> 5;
    const int wM = wave & 1;                  // 2 waves along M (64 each)
    const int wN = wave >> 1;                 // 4 waves along N (64 each)
    const int half = lane >> 4;
    const int ln16 = lane & 15;
    const int bm = blockIdx.y * BM;
    const int bn = blockIdx.x * BN;

    v8f acc[4][4];
#pragma unroll
    for (int im = 0; im < 4; ++im)
#pragma unroll
        for (int in = 0; in < 4; ++in)
#pragma unroll
            for (int r = 0; r < 8; ++r) acc[im][in][r] = 0.0f;

    const uint4* A4 = (const uint4*)A;
    const uint4* B4 = (const uint4*)Bt;
    const int K8 = K >> 3;                    // uint4 per row

    for (int k0 = 0; k0 < K; k0 += BK) {
        const int kq = k0 >> 3;
        uint4* sA4 = (uint4*)sA;
        uint4* sB4 = (uint4*)sB;
#pragma unroll
        for (int i = tid; i < BM * 8; i += 256) {           // 8 uint4 per row
            int r = i >> 3, c = i & 7;
            sA4[i] = A4[(size_t)(bm + r) * K8 + kq + c];
        }
#pragma unroll
        for (int i = tid; i < BN * 8; i += 256) {
            int r = i >> 3, c = i & 7;
            sB4[i] = B4[(size_t)(bn + r) * K8 + kq + c];
        }
        if (k0 + BK < K) {                                  // prefetch next tiles
            __builtin_prefetch(&A4[(size_t)(bm + (tid & 127)) * K8 + kq + 8], 0, 0);
            __builtin_prefetch(&B4[(size_t)(bn + tid) * K8 + kq + 8], 0, 0);
        }
        __syncthreads();

#pragma unroll
        for (int kc = 0; kc < 2; ++kc) {
            Frag af[4], bfm[4];
#pragma unroll
            for (int im = 0; im < 4; ++im) {
                int row = wM * 64 + im * 16 + ln16;
#pragma unroll
                for (int v = 0; v < 8; ++v)
                    af[im].u[v] = sA[row * 32 + kc * 16 + kpair(v, half)];
            }
#pragma unroll
            for (int in = 0; in < 4; ++in) {
                int row = wN * 64 + in * 16 + ln16;
#pragma unroll
                for (int v = 0; v < 8; ++v)
                    bfm[in].u[v] = sB[row * 32 + kc * 16 + kpair(v, half)];
            }
#pragma unroll
            for (int im = 0; im < 4; ++im)
#pragma unroll
                for (int in = 0; in < 4; ++in)
                    acc[im][in] = __builtin_amdgcn_wmma_f32_16x16x32_bf16(
                        false, af[im].v, false, bfm[in].v,
                        (short)0, acc[im][in], false, false);
        }
        __syncthreads();
    }

    // epilogue (C layout: VGPR r -> row M = r + 8*half, col N = lane%16)
#pragma unroll
    for (int im = 0; im < 4; ++im) {
#pragma unroll
        for (int in = 0; in < 4; ++in) {
            int col = bn + wN * 64 + in * 16 + ln16;
            float bv = 0.0f;
            if (EPI >= 2) bv = bias[col];
#pragma unroll
            for (int r = 0; r < 8; ++r) {
                int row = bm + wM * 64 + im * 16 + r + 8 * half;
                float v = acc[im][in][r] + bv;
                if (EPI == 3) v = gelu_tanh(v);
                if (EPI == 0 || EPI == 2)
                    ((float*)Cout)[(size_t)row * N + col] = v;
                else
                    ((unsigned short*)Cout)[(size_t)row * N + col] = f32_to_bf16(v);
            }
        }
    }
}

// ---------------------------------------------------------------------------
// Flash attention (bf16 in/out, fp32 softmax & accum).
// Q,K,V,O all (B, L, DMODEL); head h = cols [h*128, h*128+128).
// grid (Sq/128, NHEAD, BATCH), 256 threads (8 waves; wave = 16 query rows).
// K staged row-major; V staged TRANSPOSED (sVt[dh][key]) so P@V B-fragments
// are contiguous b128; P staged as bf16.
// ---------------------------------------------------------------------------
__global__ void __launch_bounds__(256)
flash_attn(const unsigned short* __restrict__ Q,
           const unsigned short* __restrict__ Kg,
           const unsigned short* __restrict__ Vg,
           unsigned short* __restrict__ Og,
           int Sq, int Skv, float scale) {
    __shared__ __align__(16) unsigned short sK[32 * DHEAD];    // [key][dh]
    __shared__ __align__(16) unsigned short sVt[DHEAD * 32];   // [dh][key]
    __shared__ __align__(16) unsigned short sPb[8][16 * 32];   // [wave][m][key]
    const int tid = threadIdx.x;
    const int lane = tid & 31;
    const int wave = tid >> 5;
    const int half = lane >> 4;
    const int ln16 = lane & 15;
    const int h = blockIdx.y;
    const int b = blockIdx.z;
    const int qbase = blockIdx.x * 128 + wave * 16;
    const int Dd = DMODEL / 2;      // dwords per row
    const int Dq = DMODEL / 8;      // uint4 per row

    const unsigned* Qu = (const unsigned*)Q;
    const uint4* K4 = (const uint4*)Kg;
    const uint4* V4 = (const uint4*)Vg;

    // Q fragments: 4 dh-chunks of 32 (A-matrix 16x32), resident for whole loop
    Frag qf[4];
    {
        int row = qbase + ln16;
        size_t base = (size_t)(b * Sq + row) * Dd + h * (DHEAD / 2);
#pragma unroll
        for (int kc = 0; kc < 4; ++kc)
#pragma unroll
            for (int v = 0; v < 8; ++v)
                qf[kc].u[v] = Qu[base + kc * 16 + kpair(v, half)];
    }

    float mstate[8], lstate[8];
    v8f o[8];
#pragma unroll
    for (int r = 0; r < 8; ++r) { mstate[r] = -1e30f; lstate[r] = 0.0f; }
#pragma unroll
    for (int n = 0; n < 8; ++n)
#pragma unroll
        for (int r = 0; r < 8; ++r) o[n][r] = 0.0f;

    const unsigned* sKu  = (const unsigned*)sK;
    const unsigned* sVtu = (const unsigned*)sVt;
    const unsigned* sPu  = (const unsigned*)&sPb[wave][0];

    for (int kv0 = 0; kv0 < Skv; kv0 += 32) {
        // --- stage K (row-major) and V (transposed) ---
        uint4* sK4 = (uint4*)sK;
#pragma unroll
        for (int e = tid; e < 512; e += 256) {        // 32 rows x 16 uint4
            int r = e >> 4, cq = e & 15;
            size_t src = (size_t)(b * Skv + kv0 + r) * Dq + h * 16 + cq;
            sK4[e] = K4[src];
            unsigned wd[4];
            *(uint4*)wd = V4[src];
#pragma unroll
            for (int j = 0; j < 4; ++j) {
                int dh = 8 * cq + 2 * j;              // dh pair base
                sVt[(dh    ) * 32 + r] = (unsigned short)(wd[j] & 0xFFFFu);
                sVt[(dh + 1) * 32 + r] = (unsigned short)(wd[j] >> 16);
            }
        }
        __syncthreads();

        // --- scores: 16 q-rows x 32 keys = two 16x16 C tiles ---
        v8f s[2];
#pragma unroll
        for (int t = 0; t < 2; ++t) {
#pragma unroll
            for (int r = 0; r < 8; ++r) s[t][r] = 0.0f;
#pragma unroll
            for (int kc = 0; kc < 4; ++kc) {
                Frag kb;
                int key = t * 16 + ln16;              // lane%16 = N (key)
#pragma unroll
                for (int v = 0; v < 8; ++v)
                    kb.u[v] = sKu[key * 64 + kc * 16 + kpair(v, half)];
                s[t] = __builtin_amdgcn_wmma_f32_16x16x32_bf16(
                    false, qf[kc].v, false, kb.v, (short)0, s[t], false, false);
            }
        }

        // --- online softmax (row = r + 8*half; reduce over lane%16 group) ---
        float alpha[8];
#pragma unroll
        for (int r = 0; r < 8; ++r) {
            float s0 = s[0][r] * scale, s1 = s[1][r] * scale;
            float mx = redmax16(fmaxf(s0, s1));
            float newm = fmaxf(mstate[r], mx);
            float a = __expf(mstate[r] - newm);
            float p0 = __expf(s0 - newm);
            float p1 = __expf(s1 - newm);
            float rs = redsum16(p0 + p1);
            lstate[r] = lstate[r] * a + rs;
            mstate[r] = newm;
            alpha[r] = a;
            int mrow = r + 8 * half;
            sPb[wave][mrow * 32 + ln16]      = f32_to_bf16(p0);
            sPb[wave][mrow * 32 + 16 + ln16] = f32_to_bf16(p1);
        }
#pragma unroll
        for (int n = 0; n < 8; ++n)
#pragma unroll
            for (int r = 0; r < 8; ++r) o[n][r] *= alpha[r];
        __syncthreads();   // P visible; all waves done with sK reads

        // --- P (16x32, A-frag from sPb) @ V (32x128 via sVt) ---
        Frag pa;
#pragma unroll
        for (int v = 0; v < 8; ++v)
            pa.u[v] = sPu[ln16 * 16 + kpair(v, half)];
#pragma unroll
        for (int n = 0; n < 8; ++n) {
            Frag vfr;
            int col = n * 16 + ln16;                  // dh col within head
#pragma unroll
            for (int v = 0; v < 8; ++v)
                vfr.u[v] = sVtu[col * 16 + kpair(v, half)];
            o[n] = __builtin_amdgcn_wmma_f32_16x16x32_bf16(
                false, pa.v, false, vfr.v, (short)0, o[n], false, false);
        }
        __syncthreads();   // done with sVt/sPb before next stage
    }

    // --- finalize: O /= l, write bf16 ---
#pragma unroll
    for (int n = 0; n < 8; ++n) {
        int col = h * DHEAD + n * 16 + ln16;
#pragma unroll
        for (int r = 0; r < 8; ++r) {
            int row = qbase + r + 8 * half;
            float val = o[n][r] / lstate[r];
            Og[(size_t)(b * Sq + row) * DMODEL + col] = f32_to_bf16(val);
        }
    }
}

// ---------------------------------------------------------------------------
// Host orchestration
// ---------------------------------------------------------------------------
extern "C" void kernel_launch(void* const* d_in, const int* in_sizes, int n_in,
                              void* d_out, int out_size, void* d_ws, size_t ws_size,
                              hipStream_t stream) {
    (void)in_sizes; (void)n_in; (void)out_size; (void)ws_size;
    const float* hid_in = (const float*)d_in[0];
    const float* enc    = (const float*)d_in[1];
    const float* temb   = (const float*)d_in[2];
    const float* rcos   = (const float*)d_in[3];
    const float* rsin   = (const float*)d_in[4];
    const int*   segp   = (const int*)d_in[5];
    const float* Wq1 = (const float*)d_in[6];
    const float* Wk1 = (const float*)d_in[7];
    const float* Wv1 = (const float*)d_in[8];
    const float* Wo1 = (const float*)d_in[9];
    const float* bo1 = (const float*)d_in[10];
    const float* Wq2 = (const float*)d_in[11];
    const float* Wk2 = (const float*)d_in[12];
    const float* Wv2 = (const float*)d_in[13];
    const float* Wo2 = (const float*)d_in[14];
    const float* bo2 = (const float*)d_in[15];
    const float* Wf1 = (const float*)d_in[16];
    const float* bf1 = (const float*)d_in[17];
    const float* Wf2 = (const float*)d_in[18];
    const float* bf2 = (const float*)d_in[19];
    const float* sst = (const float*)d_in[20];

    const size_t nDD  = (size_t)DMODEL * DMODEL;
    const size_t nDF  = (size_t)DMODEL * FFNDIM;
    const size_t nBSD = (size_t)BATCH * SEQ * DMODEL;
    const size_t nBLD = (size_t)BATCH * LENC * DMODEL;
    const size_t nBSF = (size_t)BATCH * SEQ * FFNDIM;

    char* ws = (char*)d_ws;
    size_t off = 0;
    auto alloc = [&](size_t bytes) -> void* {
        void* p = ws + off;
        off = (off + bytes + 255) & ~(size_t)255;
        return p;
    };

    // transposed bf16 weights (N x K layout)
    unsigned short* wq1t = (unsigned short*)alloc(nDD * 2);
    unsigned short* wk1t = (unsigned short*)alloc(nDD * 2);
    unsigned short* wv1t = (unsigned short*)alloc(nDD * 2);
    unsigned short* wo1t = (unsigned short*)alloc(nDD * 2);
    unsigned short* wq2t = (unsigned short*)alloc(nDD * 2);
    unsigned short* wk2t = (unsigned short*)alloc(nDD * 2);
    unsigned short* wv2t = (unsigned short*)alloc(nDD * 2);
    unsigned short* wo2t = (unsigned short*)alloc(nDD * 2);
    unsigned short* wf1t = (unsigned short*)alloc(nDF * 2);
    unsigned short* wf2t = (unsigned short*)alloc(nDF * 2);
    // activations
    unsigned short* encb  = (unsigned short*)alloc(nBLD * 2);
    unsigned short* xb    = (unsigned short*)alloc(nBSD * 2);   // LN-mod out (also y)
    float*          qf    = (float*)alloc(nBSD * 4);            // q pre-rope
    float*          kf    = (float*)alloc(nBSD * 4);            // k pre-rope
    unsigned short* qrb   = (unsigned short*)alloc(nBSD * 2);   // q roped (also q2)
    unsigned short* krb   = (unsigned short*)alloc(nBSD * 2);   // k roped (also attn2)
    unsigned short* vb    = (unsigned short*)alloc(nBSD * 2);
    unsigned short* attn1 = (unsigned short*)alloc(nBSD * 2);
    unsigned short* hidb  = (unsigned short*)alloc(nBSD * 2);
    unsigned short* k2b   = (unsigned short*)alloc(nBLD * 2);
    unsigned short* v2b   = (unsigned short*)alloc(nBLD * 2);
    float*          of    = (float*)alloc(nBSD * 4);            // GEMM f32 out
    float*          hidf  = (float*)alloc(nBSD * 4);            // running residual
    unsigned short* ff1b  = (unsigned short*)alloc(nBSF * 2);

    const int CVTB = 2048, EW = 4096;
    const dim3 blk(256);
    const float scale = 0.08838834764831845f;   // 1/sqrt(128)
    const dim3 gTDD(DMODEL / 32, DMODEL / 32);

    // --- transpose + convert weights; convert encoder states ---
    transpose_cvt<<<gTDD, blk, 0, stream>>>(Wq1, wq1t, DMODEL, DMODEL);
    transpose_cvt<<<gTDD, blk, 0, stream>>>(Wk1, wk1t, DMODEL, DMODEL);
    transpose_cvt<<<gTDD, blk, 0, stream>>>(Wv1, wv1t, DMODEL, DMODEL);
    transpose_cvt<<<gTDD, blk, 0, stream>>>(Wo1, wo1t, DMODEL, DMODEL);
    transpose_cvt<<<gTDD, blk, 0, stream>>>(Wq2, wq2t, DMODEL, DMODEL);
    transpose_cvt<<<gTDD, blk, 0, stream>>>(Wk2, wk2t, DMODEL, DMODEL);
    transpose_cvt<<<gTDD, blk, 0, stream>>>(Wv2, wv2t, DMODEL, DMODEL);
    transpose_cvt<<<gTDD, blk, 0, stream>>>(Wo2, wo2t, DMODEL, DMODEL);
    transpose_cvt<<<dim3(FFNDIM / 32, DMODEL / 32), blk, 0, stream>>>(Wf1, wf1t, DMODEL, FFNDIM);
    transpose_cvt<<<dim3(DMODEL / 32, FFNDIM / 32), blk, 0, stream>>>(Wf2, wf2t, FFNDIM, DMODEL);
    cvt_f32_bf16<<<CVTB, blk, 0, stream>>>(enc, encb, nBLD);

    // --- norm1 + modulation (shift=0, scale=1) ---
    ln_mod_kernel<<<BATCH * SEQ, blk, 0, stream>>>(hid_in, temb, sst, segp, 0, 1, xb);

    // --- self-attn projections ---
    const dim3 gDD(DMODEL / BN, (BATCH * SEQ) / BM);
    gemm_nt<0><<<gDD, blk, 0, stream>>>(xb, wq1t, nullptr, qf, BATCH * SEQ, DMODEL, DMODEL);
    gemm_nt<0><<<gDD, blk, 0, stream>>>(xb, wk1t, nullptr, kf, BATCH * SEQ, DMODEL, DMODEL);
    gemm_nt<1><<<gDD, blk, 0, stream>>>(xb, wv1t, nullptr, vb, BATCH * SEQ, DMODEL, DMODEL);

    // --- RoPE ---
    rope_bf16<<<EW, blk, 0, stream>>>(qf, rcos, rsin, qrb, nBSD / 2);
    rope_bf16<<<EW, blk, 0, stream>>>(kf, rcos, rsin, krb, nBSD / 2);

    // --- self attention ---
    flash_attn<<<dim3(SEQ / 128, NHEAD, BATCH), blk, 0, stream>>>(
        qrb, krb, vb, attn1, SEQ, SEQ, scale);

    // --- output proj + gated residual (gate idx 2) ---
    gemm_nt<2><<<gDD, blk, 0, stream>>>(attn1, wo1t, bo1, of, BATCH * SEQ, DMODEL, DMODEL);
    gate_residual<<<EW, blk, 0, stream>>>(hid_in, of, temb, sst, segp, 2, hidf, nBSD);

    // --- cross attention (x2 = hidden_states, per reference) ---
    cvt_f32_bf16<<<CVTB, blk, 0, stream>>>(hidf, hidb, nBSD);
    gemm_nt<1><<<gDD, blk, 0, stream>>>(hidb, wq2t, nullptr, qrb, BATCH * SEQ, DMODEL, DMODEL);
    const dim3 gED(DMODEL / BN, (BATCH * LENC) / BM);
    gemm_nt<1><<<gED, blk, 0, stream>>>(encb, wk2t, nullptr, k2b, BATCH * LENC, DMODEL, DMODEL);
    gemm_nt<1><<<gED, blk, 0, stream>>>(encb, wv2t, nullptr, v2b, BATCH * LENC, DMODEL, DMODEL);
    flash_attn<<<dim3(SEQ / 128, NHEAD, BATCH), blk, 0, stream>>>(
        qrb, k2b, v2b, krb /*attn2*/, SEQ, LENC, scale);
    gemm_nt<2><<<gDD, blk, 0, stream>>>(krb, wo2t, bo2, of, BATCH * SEQ, DMODEL, DMODEL);
    gate_residual<<<EW, blk, 0, stream>>>(hidf, of, temb, sst, segp, -1, hidf, nBSD);

    // --- FFN: norm3 + modulation (3,4), GEMM+GELU, GEMM, gate idx 5 ---
    ln_mod_kernel<<<BATCH * SEQ, blk, 0, stream>>>(hidf, temb, sst, segp, 3, 4, xb);
    gemm_nt<3><<<dim3(FFNDIM / BN, (BATCH * SEQ) / BM), blk, 0, stream>>>(
        xb, wf1t, bf1, ff1b, BATCH * SEQ, FFNDIM, DMODEL);
    gemm_nt<2><<<gDD, blk, 0, stream>>>(ff1b, wf2t, bf2, of, BATCH * SEQ, DMODEL, FFNDIM);
    gate_residual<<<EW, blk, 0, stream>>>(hidf, of, temb, sst, segp, 5, (float*)d_out, nBSD);
}
